// SpanQuery_39599598469525
// MI455X (gfx1250) — compile-verified, matrix operand order checked by reading, not run on hardware
//
#include <hip/hip_runtime.h>

// ---------------------------------------------------------------------------
// SpanQuery as one fused GEMM on CDNA5 (gfx1250, wave32, WMMA + async-to-LDS).
//
//   out[b,l,s,e] = relu( sum_d h[b,l,d]*q[d,s]*W[e,d] + bias[e] )
//                = relu( H[M,K] x Wf[K,N] + bias ),  Wf[d, s*768+e] = q[d,s]*W[e,d]
//   M = 8192, N = 9216, K = 768.
//
// Pass 1 (prep): split fp32 operands into bf16 hi/lo panels in d_ws
//   (x = hi + lo, ~16 mantissa bits total). ~54 MB, L2-resident.
// Pass 2 (gemm): double-buffered LDS pipeline fed by
//   GLOBAL_LOAD_ASYNC_TO_LDS_B128 (ASYNCcnt), 3x v_wmma_f32_16x16x32_bf16
//   per K=32 chunk (hi*hi + hi*lo + lo*hi), fp32 accumulate, bias+ReLU epilogue.
// Fallback (ws too small): round-1 convert-on-the-fly kernel.
// ---------------------------------------------------------------------------

typedef __bf16  bf16;
typedef bf16    bf16x8  __attribute__((ext_vector_type(8)));
typedef bf16    bf16x16 __attribute__((ext_vector_type(16)));
typedef float   f32x4   __attribute__((ext_vector_type(4)));
typedef float   f32x8   __attribute__((ext_vector_type(8)));

#define D_DIM   768
#define S_DIM   12
#define N_DIM   (S_DIM * D_DIM)   // 9216
#define M_DIM   (4 * 2048)        // 8192
#define TILE_M  128
#define TILE_N  128
#define KBLK    32                // one bf16 WMMA K-step
#define LDSS    40                // padded LDS row stride (bf16), 80B, 16B-aligned
#define KSTEPS  (D_DIM / KBLK)    // 24

#define SZ_A ((size_t)M_DIM * D_DIM * sizeof(bf16))   // 12,582,912 B
#define SZ_B ((size_t)N_DIM * D_DIM * sizeof(bf16))   // 14,155,776 B
#define WS_NEEDED (2 * (SZ_A + SZ_B))                 // ~53.5 MB

__device__ __forceinline__ void bf16_split(float x, bf16& hi, bf16& lo) {
    hi = (bf16)x;
    lo = (bf16)(x - (float)hi);
}

// --------------------------- prep kernels ----------------------------------

__global__ __launch_bounds__(256)
void prep_a_kernel(const float* __restrict__ h, bf16* __restrict__ ah,
                   bf16* __restrict__ al) {
    const size_t i = ((size_t)blockIdx.x * 256 + threadIdx.x) * 4;  // < M*K
    f32x4 g = *(const f32x4*)(h + i);
    #pragma unroll
    for (int j = 0; j < 4; ++j) bf16_split(g[j], ah[i + j], al[i + j]);
}

__global__ __launch_bounds__(256)
void prep_b_kernel(const float* __restrict__ W, const float* __restrict__ qseg,
                   bf16* __restrict__ bh, bf16* __restrict__ bl) {
    const size_t i = ((size_t)blockIdx.x * 256 + threadIdx.x) * 4;  // < N*K
    const int n = (int)(i / D_DIM);     // n = s*768 + e
    const int d = (int)(i % D_DIM);     // aligned to 4; 768 % 4 == 0
    const int s = n / D_DIM;
    const int e = n % D_DIM;
    f32x4 w = *(const f32x4*)(W + (size_t)e * D_DIM + d);
    #pragma unroll
    for (int j = 0; j < 4; ++j)
        bf16_split(w[j] * qseg[(size_t)(d + j) * S_DIM + s], bh[i + j], bl[i + j]);
}

// ------------------------- async GEMM kernel -------------------------------

__device__ __forceinline__ void async_ld_b128(unsigned lds_addr, const bf16* gaddr) {
    // GLOBAL_LOAD_ASYNC_TO_LDS_B128: VDST = LDS byte address, VADDR = 64b global.
    asm volatile("global_load_async_to_lds_b128 %0, %1, off"
                 :: "v"(lds_addr), "v"(gaddr) : "memory");
}

__global__ __launch_bounds__(256)
void gemm_async_kernel(const bf16* __restrict__ Ah, const bf16* __restrict__ Al,
                       const bf16* __restrict__ Bh, const bf16* __restrict__ Bl,
                       const float* __restrict__ bias, float* __restrict__ out) {
    // [buf][array: 0=Ahi 1=Alo 2=Bhi 3=Blo][128 rows x LDSS]
    __shared__ __align__(16) bf16 smem[2][4][TILE_M * LDSS];

    const int tid = threadIdx.x;
    const int nTiles = N_DIM / TILE_N;            // 72
    const int mT = blockIdx.x / nTiles;
    const int nT = blockIdx.x % nTiles;
    const int m0 = mT * TILE_M;
    const int n0 = nT * TILE_N;
    const int e0 = n0 % D_DIM;                    // tile lies in a single s block

    const bf16* gbase[4] = { Ah + (size_t)m0 * D_DIM, Al + (size_t)m0 * D_DIM,
                             Bh + (size_t)n0 * D_DIM, Bl + (size_t)n0 * D_DIM };

    const int wid  = tid >> 5;
    const int lane = tid & 31;
    const int half = lane >> 4;
    const int l16  = lane & 15;
    const int wm   = wid >> 1;                    // 0..3 (x32 rows)
    const int wn   = wid & 1;                     // 0..1 (x64 cols)

    f32x8 acc[2][4];
    #pragma unroll
    for (int mt = 0; mt < 2; ++mt)
        #pragma unroll
        for (int nt = 0; nt < 4; ++nt) acc[mt][nt] = (f32x8)0.0f;

    // Stage one K-step panel set: 4 arrays x 2 b128 chunks per thread
    // (128 rows x 32 bf16 = 512 x 16B chunks per array, 256 threads).
    auto stage = [&](int buf, int k0) {
        #pragma unroll
        for (int a = 0; a < 4; ++a) {
            #pragma unroll
            for (int i = 0; i < 2; ++i) {
                const int c    = tid + i * 256;
                const int row  = c >> 2;
                const int koff = (c & 3) << 3;
                const bf16* src = gbase[a] + (size_t)row * D_DIM + k0 + koff;
                unsigned lds = (unsigned)(uintptr_t)&smem[buf][a][row * LDSS + koff];
                async_ld_b128(lds, src);
            }
        }
    };

    auto loadFrag = [&](const bf16* base, int row) -> bf16x16 {
        const bf16* p = base + row * LDSS;
        bf16x8 c0 = *(const bf16x8*)(p + half * 8);
        bf16x8 c1 = *(const bf16x8*)(p + 16 + half * 8);
        return __builtin_shufflevector(c0, c1, 0, 1, 2, 3, 4, 5, 6, 7,
                                               8, 9, 10, 11, 12, 13, 14, 15);
    };

    stage(0, 0);

    #pragma unroll 1
    for (int kb = 0; kb < KSTEPS; ++kb) {
        const int buf = kb & 1;
        if (kb + 1 < KSTEPS) {
            stage(buf ^ 1, (kb + 1) * KBLK);
            // 8 asyncs just issued for the next buffer; <=8 outstanding means
            // the previous stage (this buffer) has fully landed in LDS.
            asm volatile("s_wait_asynccnt 0x8" ::: "memory");
        } else {
            asm volatile("s_wait_asynccnt 0x0" ::: "memory");
        }
        __syncthreads();

        bf16x16 aHi[2], aLo[2], bHiF[4], bLoF[4];
        #pragma unroll
        for (int mt = 0; mt < 2; ++mt) {
            const int r = wm * 32 + mt * 16 + l16;
            aHi[mt] = loadFrag(smem[buf][0], r);
            aLo[mt] = loadFrag(smem[buf][1], r);
        }
        #pragma unroll
        for (int nt = 0; nt < 4; ++nt) {
            const int r = wn * 64 + nt * 16 + l16;
            bHiF[nt] = loadFrag(smem[buf][2], r);
            bLoF[nt] = loadFrag(smem[buf][3], r);
        }
        #pragma unroll
        for (int mt = 0; mt < 2; ++mt)
            #pragma unroll
            for (int nt = 0; nt < 4; ++nt) {
                acc[mt][nt] = __builtin_amdgcn_wmma_f32_16x16x32_bf16(
                    false, aHi[mt], false, bHiF[nt], (short)0, acc[mt][nt], false, false);
                acc[mt][nt] = __builtin_amdgcn_wmma_f32_16x16x32_bf16(
                    false, aHi[mt], false, bLoF[nt], (short)0, acc[mt][nt], false, false);
                acc[mt][nt] = __builtin_amdgcn_wmma_f32_16x16x32_bf16(
                    false, aLo[mt], false, bHiF[nt], (short)0, acc[mt][nt], false, false);
            }
        __syncthreads();
    }

    // Epilogue: C layout (32-bit 16x16): VGPR r -> M = r + 8*half, lane&15 -> N.
    #pragma unroll
    for (int nt = 0; nt < 4; ++nt) {
        const int ncol = n0 + wn * 64 + nt * 16 + l16;
        const float bv = bias[e0 + wn * 64 + nt * 16 + l16];
        #pragma unroll
        for (int mt = 0; mt < 2; ++mt) {
            const int mrow0 = m0 + wm * 32 + mt * 16 + half * 8;
            #pragma unroll
            for (int r = 0; r < 8; ++r) {
                float v = acc[mt][nt][r] + bv;
                out[(size_t)(mrow0 + r) * N_DIM + ncol] = v > 0.0f ? v : 0.0f;
            }
        }
    }
}

// ---------------- fallback: convert-on-the-fly (round-1) -------------------

__global__ __launch_bounds__(256)
void spanquery_fallback_kernel(const float* __restrict__ h,
                               const float* __restrict__ qseg,
                               const float* __restrict__ W,
                               const float* __restrict__ bias,
                               float*       __restrict__ out) {
    __shared__ __align__(16) bf16 Ahi[TILE_M * LDSS];
    __shared__ __align__(16) bf16 Alo[TILE_M * LDSS];
    __shared__ __align__(16) bf16 Bhi[TILE_N * LDSS];
    __shared__ __align__(16) bf16 Blo[TILE_N * LDSS];
    __shared__ float qcol[D_DIM];

    const int tid = threadIdx.x;
    const int nTiles = N_DIM / TILE_N;
    const int mT = blockIdx.x / nTiles;
    const int nT = blockIdx.x % nTiles;
    const int m0 = mT * TILE_M;
    const int n0 = nT * TILE_N;
    const int s  = n0 / D_DIM;
    const int e0 = n0 - s * D_DIM;

    for (int i = tid; i < D_DIM; i += 256) qcol[i] = qseg[i * S_DIM + s];

    const int wid  = tid >> 5;
    const int lane = tid & 31;
    const int half = lane >> 4;
    const int l16  = lane & 15;
    const int wm   = wid >> 1;
    const int wn   = wid & 1;

    f32x8 acc[2][4];
    for (int mt = 0; mt < 2; ++mt)
        for (int nt = 0; nt < 4; ++nt) acc[mt][nt] = (f32x8)0.0f;

    auto loadFrag = [&](const bf16* base, int row) -> bf16x16 {
        const bf16* p = base + row * LDSS;
        bf16x8 c0 = *(const bf16x8*)(p + half * 8);
        bf16x8 c1 = *(const bf16x8*)(p + 16 + half * 8);
        return __builtin_shufflevector(c0, c1, 0, 1, 2, 3, 4, 5, 6, 7,
                                               8, 9, 10, 11, 12, 13, 14, 15);
    };

    for (int k0 = 0; k0 < D_DIM; k0 += KBLK) {
        __syncthreads();
        for (int c = tid; c < (TILE_M * KBLK / 4); c += 256) {
            const int row = c >> 3, kq = (c & 7) << 2;
            f32x4 g = *(const f32x4*)(h + (size_t)(m0 + row) * D_DIM + k0 + kq);
            #pragma unroll
            for (int j = 0; j < 4; ++j)
                bf16_split(g[j], Ahi[row * LDSS + kq + j], Alo[row * LDSS + kq + j]);
        }
        for (int c = tid; c < (TILE_N * KBLK / 4); c += 256) {
            const int row = c >> 3, kq = (c & 7) << 2;
            f32x4 g = *(const f32x4*)(W + (size_t)(e0 + row) * D_DIM + k0 + kq);
            #pragma unroll
            for (int j = 0; j < 4; ++j)
                bf16_split(g[j] * qcol[k0 + kq + j],
                           Bhi[row * LDSS + kq + j], Blo[row * LDSS + kq + j]);
        }
        __syncthreads();

        bf16x16 aHi[2], aLo[2], bHiF[4], bLoF[4];
        #pragma unroll
        for (int mt = 0; mt < 2; ++mt) {
            const int r = wm * 32 + mt * 16 + l16;
            aHi[mt] = loadFrag(Ahi, r);
            aLo[mt] = loadFrag(Alo, r);
        }
        #pragma unroll
        for (int nt = 0; nt < 4; ++nt) {
            const int r = wn * 64 + nt * 16 + l16;
            bHiF[nt] = loadFrag(Bhi, r);
            bLoF[nt] = loadFrag(Blo, r);
        }
        #pragma unroll
        for (int mt = 0; mt < 2; ++mt)
            #pragma unroll
            for (int nt = 0; nt < 4; ++nt) {
                acc[mt][nt] = __builtin_amdgcn_wmma_f32_16x16x32_bf16(
                    false, aHi[mt], false, bHiF[nt], (short)0, acc[mt][nt], false, false);
                acc[mt][nt] = __builtin_amdgcn_wmma_f32_16x16x32_bf16(
                    false, aHi[mt], false, bLoF[nt], (short)0, acc[mt][nt], false, false);
                acc[mt][nt] = __builtin_amdgcn_wmma_f32_16x16x32_bf16(
                    false, aLo[mt], false, bHiF[nt], (short)0, acc[mt][nt], false, false);
            }
    }

    #pragma unroll
    for (int nt = 0; nt < 4; ++nt) {
        const int ncol = n0 + wn * 64 + nt * 16 + l16;
        const float bv = bias[e0 + wn * 64 + nt * 16 + l16];
        #pragma unroll
        for (int mt = 0; mt < 2; ++mt) {
            const int mrow0 = m0 + wm * 32 + mt * 16 + half * 8;
            #pragma unroll
            for (int r = 0; r < 8; ++r) {
                float v = acc[mt][nt][r] + bv;
                out[(size_t)(mrow0 + r) * N_DIM + ncol] = v > 0.0f ? v : 0.0f;
            }
        }
    }
}

// ------------------------------ launch -------------------------------------

extern "C" void kernel_launch(void* const* d_in, const int* in_sizes, int n_in,
                              void* d_out, int out_size, void* d_ws, size_t ws_size,
                              hipStream_t stream) {
    const float* h    = (const float*)d_in[0];   // (B,L,D)
    const float* qseg = (const float*)d_in[1];   // (D,S)
    const float* W    = (const float*)d_in[2];   // (D,D) [e,d]
    const float* bias = (const float*)d_in[3];   // (D,)
    float* out        = (float*)d_out;

    const int grid = (M_DIM / TILE_M) * (N_DIM / TILE_N);  // 4608

    if (ws_size >= WS_NEEDED) {
        bf16* Ah = (bf16*)d_ws;
        bf16* Al = (bf16*)((char*)d_ws + SZ_A);
        bf16* Bh = (bf16*)((char*)d_ws + 2 * SZ_A);
        bf16* Bl = (bf16*)((char*)d_ws + 2 * SZ_A + SZ_B);
        prep_a_kernel<<<(M_DIM * D_DIM) / 1024, 256, 0, stream>>>(h, Ah, Al);
        prep_b_kernel<<<(N_DIM * D_DIM) / 1024, 256, 0, stream>>>(W, qseg, Bh, Bl);
        gemm_async_kernel<<<grid, 256, 0, stream>>>(Ah, Al, Bh, Bl, bias, out);
    } else {
        spanquery_fallback_kernel<<<grid, 256, 0, stream>>>(h, qseg, W, bias, out);
    }
}
